// ModelNERMulti_17987323036291
// MI455X (gfx1250) — compile-verified, matrix operand order checked by reading, not run on hardware
//
#include <hip/hip_runtime.h>
#include <hip/hip_bf16.h>

// ---- CDNA5 WMMA vector types ----
typedef __attribute__((ext_vector_type(16))) __bf16          v16bf;
typedef __attribute__((ext_vector_type(8)))  float           v8f;
typedef __attribute__((ext_vector_type(8)))  unsigned short  u16x8;

#define E_     11
#define ALPHA_ 1000.0f
#define C1_    8.0f
#define B_     16
#define S_     256
#define H_     1024
#define L_     32

#define MT 64      // M rows per block
#define NC 64      // n1 chunk (cols of W1 / rows of W2)
#define KC 64      // k slab for W1 (small -> cheap double buffering)
#define NSLAB 256  // total W1 slabs per block: (H/NC)*(H/KC)

// padded LDS pitches (elements); multiples of 8 -> 16B-aligned rows, bank-spread
#define XS_LD 1032
#define WT_LD 72
#define HB_LD 72

// LDS carve (elements of ushort)
#define XS_ELEMS   (MT * XS_LD)                 // 66048
#define W1T_ELEMS  (NC * WT_LD)                 // 4608 per buffer
#define W1T_OFF    XS_ELEMS
#define HB_OFF     (W1T_OFF + 2 * W1T_ELEMS)
#define SMEM_ELEMS (HB_OFF + MT * HB_LD)        // 79872 elems = 159744 B

// fp32 -> bf16, round-to-nearest (ties up): one add, then take high 16 bits.
// The 2^-8 bf16 quantization step dominates; ties-up vs RNE is noise here.
__device__ __forceinline__ unsigned short f2bf(float f) {
  return (unsigned short)((__float_as_uint(f) + 0x8000u) >> 16);
}

// packed fp32x2 -> bf16x2: two v_add + one v_perm_b32 (byte-select the two
// high half-words) instead of the ~20-op manual RNE chain.
__device__ __forceinline__ unsigned int f2bf_pk(float a, float b) {
  unsigned int ua = __float_as_uint(a) + 0x8000u;
  unsigned int ub = __float_as_uint(b) + 0x8000u;
#if __has_builtin(__builtin_amdgcn_perm)
  // {S0=ub -> bytes 4..7, S1=ua -> bytes 0..3}; sel 0x07060302:
  // D = [ua.b2, ua.b3, ub.b2, ub.b3] = lo16 = hi(ua), hi16 = hi(ub)
  return __builtin_amdgcn_perm(ub, ua, 0x07060302u);
#else
  return (ua >> 16) | (ub & 0xFFFF0000u);
#endif
}

// Workgroup barrier that orders ONLY LDS traffic (DScnt), leaving in-flight
// global prefetch loads (LOADcnt) untouched.  __syncthreads() would emit
// s_wait_loadcnt_dscnt 0x0 and drain the software pipeline.
__device__ __forceinline__ void lds_barrier() {
  asm volatile("s_wait_dscnt 0x0\n\t"
               "s_barrier_signal -1\n\t"
               "s_barrier_wait -1" ::: "memory");
}

// 16x32 bf16 A-fragment (or mirrored B-fragment) from row-major LDS [rows][ldk]:
// lane L (0-15) -> row row0+L, K = k0+0..7 / k0+16..23; lane L+16 -> K = +8.. / +24..
__device__ __forceinline__ v16bf lds_frag(const unsigned short* base, int row0,
                                          int ldk, int k0) {
  const int lane = threadIdx.x & 31;
  const int r    = row0 + (lane & 15);
  const int koff = k0 + ((lane >> 4) << 3);
  const unsigned short* p = base + r * ldk + koff;
  union { u16x8 h[2]; v16bf v; } u;
  u.h[0] = *(const u16x8*)p;
  u.h[1] = *(const u16x8*)(p + 16);
  return u.v;
}

// ---- W1^T slab pipeline helpers (constant-indexed -> stays in VGPRs) ----
__device__ __forceinline__ void issue_slab(const float* __restrict__ W1e, int l,
                                           int kbase, int n4, float4 (&pf)[4]) {
  if (l >= NSLAB) return;
  const int snc = (l >> 4) << 6;   // n1 chunk base
  const int skc = (l & 15) << 6;   // k slab base
#pragma unroll
  for (int i = 0; i < 4; ++i) {
    int k = kbase + i * 16;
    pf[i] = *(const float4*)(W1e + (size_t)(skc + k) * H_ + snc + n4);
  }
}

__device__ __forceinline__ void store_slab(unsigned short* buf, int kbase, int n4,
                                           const float4 (&pf)[4]) {
#pragma unroll
  for (int i = 0; i < 4; ++i) {
    int k = kbase + i * 16;
    unsigned int p01 = f2bf_pk(pf[i].x, pf[i].y);
    unsigned int p23 = f2bf_pk(pf[i].z, pf[i].w);
    buf[(n4 + 0) * WT_LD + k] = (unsigned short)p01;          // ds_store_b16
    buf[(n4 + 1) * WT_LD + k] = (unsigned short)(p01 >> 16);  // ds_store_b16_d16_hi
    buf[(n4 + 2) * WT_LD + k] = (unsigned short)p23;
    buf[(n4 + 3) * WT_LD + k] = (unsigned short)(p23 >> 16);
  }
}

__device__ __forceinline__ void gemm1_slab(const unsigned short* xs,
                                           const unsigned short* buf, int kc,
                                           int mt, int ntp, v8f& hacc0, v8f& hacc1) {
#pragma unroll
  for (int kk = 0; kk < KC; kk += 32) {
    v16bf fa  = lds_frag(xs,  mt * 16,        XS_LD, kc + kk);
    v16bf fb0 = lds_frag(buf, (ntp + 0) * 16, WT_LD, kk);
    v16bf fb1 = lds_frag(buf, (ntp + 1) * 16, WT_LD, kk);
    hacc0 = __builtin_amdgcn_wmma_f32_16x16x32_bf16(false, fa, false, fb0,
                                                    (short)0, hacc0, false, false);
    hacc1 = __builtin_amdgcn_wmma_f32_16x16x32_bf16(false, fa, false, fb1,
                                                    (short)0, hacc1, false, false);
  }
}

// ---- gate kernel: soft-argmax lang id + top-hat gates -> ws[E][B] ----
__global__ void gate_kernel(const float* __restrict__ logits,
                            float* __restrict__ gate_ws) {
  int b = threadIdx.x;
  if (b >= B_) return;
  float m = -1e30f;
  for (int e = 0; e < E_; ++e) m = fmaxf(m, logits[b * E_ + e]);
  float s = 0.f, num = 0.f;
  for (int e = 0; e < E_; ++e) {
    float w = __expf((logits[b * E_ + e] - m) * ALPHA_);
    s += w; num += w * (float)e;
  }
  float id = num / s;
  for (int e = 0; e < E_; ++e) {
    float z1 = (id - ((float)e - 0.5f)) * C1_;
    float z2 = (((float)e + 0.5f) - id) * C1_;
    float g  = (1.f / (1.f + __expf(-z1))) * (1.f / (1.f + __expf(-z2)));
    gate_ws[e * B_ + b] = g;
  }
}

// ---- fused expert FFN: out[b,s,:] += gate_e * (relu(x W1_e + b1_e) W2_e + b2_e) ----
__global__ __launch_bounds__(256) void moe_ffn_kernel(
    const float* __restrict__ x,  const float* __restrict__ W1,
    const float* __restrict__ B1, const float* __restrict__ W2,
    const float* __restrict__ B2, const float* __restrict__ gate_ws,
    float* __restrict__ out) {
  __shared__ unsigned short smem[SMEM_ELEMS];
  unsigned short* xs   = smem;                      // x tile, bf16
  unsigned short* w1t0 = smem + W1T_OFF;            // W1^T slab, double buffered
  unsigned short* w1t1 = w1t0 + W1T_ELEMS;
  unsigned short* w2t  = w1t0;                      // W2^T chunk reuses buffer 0
  unsigned short* hbuf = smem + HB_OFF;             // relu(h) chunk, bf16

  const int e  = blockIdx.y;
  const int m0 = blockIdx.x * MT;        // MT divides S -> single batch sample
  const int b  = m0 / S_;
  const float gate = gate_ws[e * B_ + b];
  if (gate < 1e-12f) return;             // contribution below fp32 accumulation ulp

  const int tid   = threadIdx.x;
  const int wave  = tid >> 5;
  const int lane  = tid & 31;
  const int col16 = lane & 15;
  const int hi8   = (lane >> 4) << 3;    // C-frag: upper half-wave holds rows +8

  const float* W1e = W1 + (size_t)e * H_ * H_;
  const float* W2e = W2 + (size_t)e * H_ * L_;

  const int n4    = (tid & 15) << 2;     // 4 consecutive n per thread
  const int kbase = tid >> 4;            // k stride 16, 4 rounds covers KC=64

  float4 pf0[4], pf1[4];                 // depth-2 pipeline, compile-time parity
  issue_slab(W1e, 0, kbase, n4, pf0);    // two slabs in flight during x staging
  issue_slab(W1e, 1, kbase, n4, pf1);

  // stage x tile [MT x H] as bf16 (coalesced float4 reads, 8B LDS stores)
  {
    const float* xrow = x + (size_t)m0 * H_;
    for (int idx = tid; idx < MT * (H_ / 4); idx += 256) {
      int rr  = idx / (H_ / 4);
      int cc4 = (idx % (H_ / 4)) * 4;
      float4 v = *(const float4*)(xrow + rr * H_ + cc4);
      unsigned long long q = (unsigned long long)f2bf_pk(v.x, v.y)
                           | ((unsigned long long)f2bf_pk(v.z, v.w) << 32);
      *(unsigned long long*)&xs[rr * XS_LD + cc4] = q;
    }
  }

  // wave tiling: GEMM1 -> wave owns (mt, ntp..ntp+1); GEMM2 -> wave owns (mt, nt2)
  const int mt  = wave & 3;
  const int ntp = (wave >> 2) << 1;
  const int nt2 = wave >> 2;

  v8f tacc = {};                          // persistent 16x16 f32 t-tile

  for (int nc = 0; nc < H_; nc += NC) {
    v8f hacc0 = {}, hacc1 = {};
    const int lch = (nc >> 6) << 4;       // first slab index of this chunk

    for (int kc = 0; kc < H_; kc += 2 * KC) {   // two slabs per iteration
      const int l = lch + (kc >> 6);
      // ---- slab A: pf0 -> w1t0 ----
      store_slab(w1t0, kbase, n4, pf0);   // s_wait_loadcnt lands here (~2 slabs cover)
      issue_slab(W1e, l + 2, kbase, n4, pf0);
      lds_barrier();                      // DScnt-only: prefetches stay in flight
      gemm1_slab(xs, w1t0, kc, mt, ntp, hacc0, hacc1);
      // ---- slab B: pf1 -> w1t1 ----
      store_slab(w1t1, kbase, n4, pf1);   // safe: all waves' w1t1 reads fenced by
      issue_slab(W1e, l + 3, kbase, n4, pf1);  // slab A's barrier
      lds_barrier();
      gemm1_slab(xs, w1t1, kc + KC, mt, ntp, hacc0, hacc1);
    }

    // chunk epilogue: bias + ReLU -> hbuf (C-frag -> A-frag layout via LDS)
    {
      float bias0 = B1[(size_t)e * H_ + nc + (ntp + 0) * 16 + col16];
      float bias1 = B1[(size_t)e * H_ + nc + (ntp + 1) * 16 + col16];
#pragma unroll
      for (int r = 0; r < 8; ++r) {
        int row = mt * 16 + r + hi8;
        unsigned int hp = f2bf_pk(fmaxf(hacc0[r] + bias0, 0.f),
                                  fmaxf(hacc1[r] + bias1, 0.f));
        hbuf[row * HB_LD + (ntp + 0) * 16 + col16] = (unsigned short)hp;
        hbuf[row * HB_LD + (ntp + 1) * 16 + col16] = (unsigned short)(hp >> 16);
      }
    }
    // stage W2^T chunk into (now idle) w1t buffer 0: w2t[n][k] = W2e[nc+k][n]
    // (next chunk's slab-0 store into this region is fenced by the trailing barrier)
    for (int i = tid; i < NC * L_; i += 256) {
      int k = i >> 5, n = i & 31;
      w2t[n * WT_LD + k] = f2bf(W2e[(size_t)(nc + k) * L_ + n]);
    }
    lds_barrier();

    // GEMM2: tacc += h_chunk[64x64] @ W2^T chunk
#pragma unroll
    for (int kk = 0; kk < NC; kk += 32) {
      v16bf fa = lds_frag(hbuf, mt * 16,  HB_LD, kk);
      v16bf fb = lds_frag(w2t,  nt2 * 16, WT_LD, kk);
      tacc = __builtin_amdgcn_wmma_f32_16x16x32_bf16(false, fa, false, fb,
                                                     (short)0, tacc, false, false);
    }
    lds_barrier();   // fences hbuf + w2t + w1t0 before next chunk reuses them
  }

  // epilogue: out += gate * (t + b2), accumulated across expert blocks
  {
    int col = nt2 * 16 + col16;
    float bias2 = B2[(size_t)e * L_ + col];
#pragma unroll
    for (int r = 0; r < 8; ++r) {
      int row = m0 + mt * 16 + r + hi8;
      atomicAdd(&out[(size_t)row * L_ + col], gate * (tacc[r] + bias2));
    }
  }
}

extern "C" void kernel_launch(void* const* d_in, const int* in_sizes, int n_in,
                              void* d_out, int out_size, void* d_ws, size_t ws_size,
                              hipStream_t stream) {
  const float* x  = (const float*)d_in[0];   // [B,S,H]
  const float* ld = (const float*)d_in[1];   // [B,E]
  const float* W1 = (const float*)d_in[2];   // [E,H,H]
  const float* b1 = (const float*)d_in[3];   // [E,H]
  const float* W2 = (const float*)d_in[4];   // [E,H,L]
  const float* b2 = (const float*)d_in[5];   // [E,L]
  float* out     = (float*)d_out;            // [B,S,L]
  float* gate_ws = (float*)d_ws;             // [E,B]

  gate_kernel<<<1, 32, 0, stream>>>(ld, gate_ws);
  hipMemsetAsync(d_out, 0, (size_t)out_size * sizeof(float), stream);

  dim3 grid((B_ * S_) / MT, E_);
  moe_ffn_kernel<<<grid, 256, 0, stream>>>(x, W1, b1, W2, b2, gate_ws, out);
}
